// XCAttention_850403524680
// MI455X (gfx1250) — compile-verified, matrix operand order checked by reading, not run on hardware
//
#include <hip/hip_runtime.h>
#include <hip/hip_bf16.h>
#include <math.h>
#include <stdint.h>

typedef __attribute__((ext_vector_type(16))) _Float16 v16h;
typedef __attribute__((ext_vector_type(8)))  float    v8f;
typedef __attribute__((ext_vector_type(4)))  unsigned u32x4;
typedef __attribute__((ext_vector_type(8)))  _Float16 h8v;
typedef __attribute__((ext_vector_type(4)))  float    f32x4;
typedef __attribute__((ext_vector_type(4)))  int      i32x4;

union FragH { v16h h; u32x4 q[2]; };

#define LN_EPS   1e-5f
#define L2_EPS   1e-12f
#define SM_SCALE 8.0f

// ---------------------------------------------------------------------------
// CDNA5 async global->LDS staging (ASYNCcnt path), with safe fallback.
// Builtin signature (leaked by compiler diagnostic): param0 = int4 addrspace(1)*.
// ---------------------------------------------------------------------------
#if defined(__HIP_DEVICE_COMPILE__) && __has_builtin(__builtin_amdgcn_global_load_async_to_lds_b128)
#define USE_ASYNC_LDS 1
#else
#define USE_ASYNC_LDS 0
#endif

#if USE_ASYNC_LDS
typedef __attribute__((address_space(1))) i32x4 g_i32x4;
typedef __attribute__((address_space(3))) i32x4 l_i32x4;
#endif

__device__ __forceinline__ void stage16(const _Float16* g, _Float16* l) {
#if USE_ASYNC_LDS
  __builtin_amdgcn_global_load_async_to_lds_b128(
      (g_i32x4*)(uintptr_t)g,
      (l_i32x4*)(uint32_t)(uintptr_t)l, 0, 0);
#else
  *(u32x4*)l = *(const u32x4*)g;
#endif
}

__device__ __forceinline__ void stage_wait() {
#if USE_ASYNC_LDS
#if __has_builtin(__builtin_amdgcn_s_wait_asynccnt)
  __builtin_amdgcn_s_wait_asynccnt(0);
#else
  asm volatile("s_wait_asynccnt 0" ::: "memory");
#endif
#endif
}

// ---------------------------------------------------------------------------
// Kernel 1: LayerNorm over channels with LDS transpose.
// x is [b][c=256][pos=4096]; we need per-pos stats over c. Load a 256c x 32pos
// tile coalesced along pos, reduce per pos, emit f16 rows t16[(b*4096+pos)][c].
// ---------------------------------------------------------------------------
__global__ __launch_bounds__(256) void k_ln(const float* __restrict__ x,
                                            const float* __restrict__ gamma,
                                            const float* __restrict__ beta,
                                            _Float16* __restrict__ t16) {
  __shared__ float xs[256 * 33];
  __shared__ float redS[256], redQ[256];
  __shared__ float muS[32], riS[32];
  const int b    = blockIdx.x;
  const int pos0 = blockIdx.y * 32;
  const int tid  = threadIdx.x;
  for (int idx = tid; idx < 256 * 32; idx += 256) {
    int c = idx >> 5, p = idx & 31;
    xs[c * 33 + p] = x[((size_t)b * 256 + c) * 4096 + pos0 + p];
  }
  __syncthreads();
  const int pl = tid & 31, sl = tid >> 5;
  float s = 0.f, q = 0.f;
  for (int c = sl * 32; c < sl * 32 + 32; ++c) {
    float v = xs[c * 33 + pl];
    s += v; q += v * v;
  }
  redS[tid] = s; redQ[tid] = q;
  __syncthreads();
  if (tid < 32) {
    float ts = 0.f, tq = 0.f;
    for (int w = 0; w < 8; ++w) { ts += redS[w * 32 + tid]; tq += redQ[w * 32 + tid]; }
    float mu  = ts * (1.0f / 256.0f);
    float var = tq * (1.0f / 256.0f) - mu * mu;
    muS[tid] = mu;
    riS[tid] = rsqrtf(var + LN_EPS);
  }
  __syncthreads();
  const float gma = gamma[tid], bta = beta[tid];
  for (int p = 0; p < 32; ++p) {
    float v = (xs[tid * 33 + p] - muS[p]) * riS[p] * gma + bta;
    t16[((size_t)b * 4096 + pos0 + p) * 256 + tid] = (_Float16)v;
  }
}

// ---------------------------------------------------------------------------
// Kernel 2: convert weights to f16, stored transposed (w16[col][k]).
// ---------------------------------------------------------------------------
__global__ __launch_bounds__(256) void k_cvtw(const float* __restrict__ Wqkv,
                                              const float* __restrict__ Wout,
                                              _Float16* __restrict__ w16,
                                              _Float16* __restrict__ wo16) {
  int idx = blockIdx.x * 256 + threadIdx.x;
  if (idx < 768 * 256) {
    int col = idx >> 8, k = idx & 255;
    w16[idx] = (_Float16)Wqkv[(size_t)k * 768 + col];
  } else {
    int i2 = idx - 768 * 256;
    if (i2 < 256 * 256) {
      int col = i2 >> 8, k = i2 & 255;
      wo16[i2] = (_Float16)Wout[(size_t)k * 256 + col];
    }
  }
}

// ---------------------------------------------------------------------------
// Kernel 3: qkv projection GEMM  [65536 x 256] x [256 x 768] with WMMA.
// Block = 128 rows x 64 cols, 8 waves; wave w computes 16x64 (4 accumulators,
// A fragment reused 4x). K steps of 32. Async global->LDS tile staging.
// Fragment layout per CDNA5 ISA 7.12.2 (16-bit A 16x32): lane l holds row
// m=l&15; halves 0..7 = K {g*8..g*8+7}, halves 8..15 = K {16+g*8..}, g=l>>4.
// Epilogue: stage through LDS, store planar qkv16[3][b][h][d][n] f16.
// ---------------------------------------------------------------------------
__global__ __launch_bounds__(256) void k_gemm_qkv(const _Float16* __restrict__ t16,
                                                  const _Float16* __restrict__ w16,
                                                  _Float16* __restrict__ qkv16) {
  __shared__ _Float16 ldsA[128 * 40];
  __shared__ _Float16 ldsB[64 * 40];
  __shared__ float    ldsC[8 * 16 * 17];
  const int tid  = threadIdx.x;
  const int wave = tid >> 5, lane = tid & 31;
  const int g = lane >> 4, lm = lane & 15;
  const int m0 = blockIdx.x * 128;
  const int n0 = blockIdx.y * 64;
  v8f acc[4] = {};
  const int arow = tid >> 1, aoff = (tid & 1) * 16;
  const _Float16* asrc = t16 + (size_t)(m0 + arow) * 256 + aoff;
  _Float16* adst = ldsA + arow * 40 + aoff;
  const int brow = tid >> 2, boff = (tid & 3) * 8;
  const _Float16* bsrc = w16 + (size_t)(n0 + brow) * 256 + boff;
  _Float16* bdst = ldsB + brow * 40 + boff;
  for (int k0 = 0; k0 < 256; k0 += 32) {
    __syncthreads();
    stage16(asrc + k0,     adst);
    stage16(asrc + k0 + 8, adst + 8);
    stage16(bsrc + k0,     bdst);
    stage_wait();
    __syncthreads();
    FragH a;
    const _Float16* ar = ldsA + (wave * 16 + lm) * 40;
    a.q[0] = *(const u32x4*)(ar + g * 8);
    a.q[1] = *(const u32x4*)(ar + 16 + g * 8);
#pragma unroll
    for (int tn = 0; tn < 4; ++tn) {
      FragH bf;
      const _Float16* br = ldsB + (tn * 16 + lm) * 40;
      bf.q[0] = *(const u32x4*)(br + g * 8);
      bf.q[1] = *(const u32x4*)(br + 16 + g * 8);
      acc[tn] = __builtin_amdgcn_wmma_f32_16x16x32_f16(false, a.h, false, bf.h,
                                                       (short)0, acc[tn], false, false);
    }
  }
  // epilogue: per N-tile, stage C via LDS, store planar, 16B runs along pos
  const int bb   = m0 >> 12;
  const int posB = m0 & 4095;
  const int rowId = tid >> 1;   // 0..127 = wave*16 + d
  const int wsrc  = rowId >> 4;
  const int d     = rowId & 15;
  const int ho    = (tid & 1) * 8;
#pragma unroll
  for (int tn = 0; tn < 4; ++tn) {
    __syncthreads();
    float* cw = ldsC + wave * (16 * 17);
#pragma unroll
    for (int r = 0; r < 8; ++r)
      cw[lm * 17 + g * 8 + r] = acc[tn][r];   // ldsC[wave][ncol=d][m=pos]
    __syncthreads();
    const int col0   = n0 + tn * 16;
    const int qkvIdx = col0 >> 8;
    const int hh     = (col0 >> 5) & 7;
    const int dbase  = col0 & 31;
    const float* csrc = ldsC + wsrc * (16 * 17) + d * 17 + ho;
    h8v ov;
#pragma unroll
    for (int e = 0; e < 8; ++e) ov[e] = (_Float16)csrc[e];
    size_t dst = ((((size_t)qkvIdx * 16 + bb) * 8 + hh) * 32 + (dbase + d)) * 4096
               + posB + wsrc * 16 + ho;
    *(h8v*)(qkv16 + dst) = ov;
  }
}

// ---------------------------------------------------------------------------
// Kernel 4: L2-normalize q and k rows (length n=4096) in place;
// q additionally scaled by exp(temperature[h]).
// ---------------------------------------------------------------------------
__global__ __launch_bounds__(256) void k_l2scale(_Float16* __restrict__ qkv16,
                                                 const float* __restrict__ temperature) {
  __shared__ float red[256];
  const int rowId = blockIdx.x;            // 0..8191
  const int qk = rowId >> 12;              // 0=q, 1=k
  const int rr = rowId & 4095;
  const int b = rr >> 8, h = (rr >> 5) & 7, d = rr & 31;
  _Float16* row = qkv16 + ((((size_t)qk * 16 + b) * 8 + h) * 32 + d) * 4096;
  const int tid = threadIdx.x;
  float v[16];
  float s = 0.f;
#pragma unroll
  for (int j = 0; j < 16; ++j) {
    v[j] = (float)row[tid + j * 256];
    s += v[j] * v[j];
  }
  red[tid] = s;
  __syncthreads();
  for (int off = 128; off > 0; off >>= 1) {
    if (tid < off) red[tid] += red[tid + off];
    __syncthreads();
  }
  float scale = 1.0f / fmaxf(sqrtf(red[0]), L2_EPS);
  if (qk == 0) scale *= __expf(temperature[h]);
#pragma unroll
  for (int j = 0; j < 16; ++j)
    row[tid + j * 256] = (_Float16)(v[j] * scale);
}

// ---------------------------------------------------------------------------
// Kernel 5: cross-covariance attention per (b,h).
//   sim = q . k^T over n (K=4096, WMMA accumulation, both operands row-major)
//   softmax(sim * 8) over j
//   out = attn . v  (K=32, one WMMA per 16x16 tile; v transposed via LDS)
// Output rows out16[(b*4096+pos)][h*32+i] f16 for the final GEMM.
// ---------------------------------------------------------------------------
__global__ __launch_bounds__(128) void k_attn(const _Float16* __restrict__ qkv16,
                                              _Float16* __restrict__ out16) {
  __shared__ float    sim[32 * 33];
  __shared__ _Float16 attnS[32 * 32];
  __shared__ _Float16 ldsV[64 * 40];
  const int bh = blockIdx.x;
  const int b = bh >> 3, h = bh & 7;
  const int tid  = threadIdx.x;
  const int wave = tid >> 5, lane = tid & 31;
  const int g = lane >> 4, lm = lane & 15;
  const size_t planeQ = (((size_t)0 * 16 + b) * 8 + h) * (size_t)32 * 4096;
  const size_t planeK = (((size_t)1 * 16 + b) * 8 + h) * (size_t)32 * 4096;
  const size_t planeV = (((size_t)2 * 16 + b) * 8 + h) * (size_t)32 * 4096;
  // ---- phase 1: sim tiles (wave -> (i_tile, j_tile))
  {
    const int it = wave & 1, jt = wave >> 1;
    const _Float16* qr = qkv16 + planeQ + (size_t)(it * 16 + lm) * 4096;
    const _Float16* kr = qkv16 + planeK + (size_t)(jt * 16 + lm) * 4096;
    v8f acc = {};
    for (int kk = 0; kk < 4096; kk += 32) {
      __builtin_prefetch((const void*)(qr + kk + 256), 0, 1);
      __builtin_prefetch((const void*)(kr + kk + 256), 0, 1);
      FragH a, bf;
      a.q[0]  = *(const u32x4*)(qr + kk + g * 8);
      a.q[1]  = *(const u32x4*)(qr + kk + 16 + g * 8);
      bf.q[0] = *(const u32x4*)(kr + kk + g * 8);
      bf.q[1] = *(const u32x4*)(kr + kk + 16 + g * 8);
      acc = __builtin_amdgcn_wmma_f32_16x16x32_f16(false, a.h, false, bf.h,
                                                   (short)0, acc, false, false);
    }
#pragma unroll
    for (int r = 0; r < 8; ++r)
      sim[(it * 16 + g * 8 + r) * 33 + jt * 16 + lm] = acc[r];
  }
  __syncthreads();
  // ---- phase 2: softmax rows (32 rows of 32)
  if (tid < 32) {
    float* row = sim + tid * 33;
    float mx = -1e30f;
    for (int j = 0; j < 32; ++j) mx = fmaxf(mx, row[j]);
    float sum = 0.f;
    for (int j = 0; j < 32; ++j) {
      float e = __expf((row[j] - mx) * SM_SCALE);
      row[j] = e; sum += e;
    }
    float inv = 1.0f / sum;
    for (int j = 0; j < 32; ++j)
      attnS[tid * 32 + j] = (_Float16)(row[j] * inv);
  }
  __syncthreads();
  // ---- phase 3: out = attn @ v  (wave -> (i_tile, pos pair))
  const int it = wave & 1, pp = wave >> 1;
  FragH a;
  {
    const _Float16* ar = attnS + (it * 16 + lm) * 32;
    a.q[0] = *(const u32x4*)(ar + g * 8);
    a.q[1] = *(const u32x4*)(ar + 16 + g * 8);
  }
  for (int pos0 = 0; pos0 < 4096; pos0 += 64) {
    __syncthreads();
    {
      // transpose v[j][pos] tile (32 x 64) into ldsV[pos][j]
      const int j = tid >> 2, ch = tid & 3;
      union { u32x4 q[2]; _Float16 t[16]; } tmp;
      const _Float16* src = qkv16 + planeV + (size_t)j * 4096 + pos0 + ch * 16;
      tmp.q[0] = *(const u32x4*)(src);
      tmp.q[1] = *(const u32x4*)(src + 8);
#pragma unroll
      for (int e = 0; e < 16; ++e)
        ldsV[(ch * 16 + e) * 40 + j] = tmp.t[e];
    }
    __syncthreads();
#pragma unroll
    for (int u = 0; u < 2; ++u) {
      const int pt = pp * 2 + u;
      FragH bf;
      const _Float16* vr = ldsV + (pt * 16 + lm) * 40;
      bf.q[0] = *(const u32x4*)(vr + g * 8);
      bf.q[1] = *(const u32x4*)(vr + 16 + g * 8);
      v8f z = {};
      v8f c = __builtin_amdgcn_wmma_f32_16x16x32_f16(false, a.h, false, bf.h,
                                                     (short)0, z, false, false);
      const int pos = pos0 + pt * 16 + lm;
      _Float16* dst = out16 + ((size_t)b * 4096 + pos) * 256 + h * 32 + it * 16 + g * 8;
      h8v ov;
#pragma unroll
      for (int r = 0; r < 8; ++r) ov[r] = (_Float16)c[r];
      *(h8v*)dst = ov;
    }
  }
}

// ---------------------------------------------------------------------------
// Kernel 6: output projection GEMM [65536 x 256] x [256 x 256] + bias,
// 128x64 block tiling, stored straight into [b][c][H][W] f32 via LDS-staged
// transpose epilogue.
// ---------------------------------------------------------------------------
__global__ __launch_bounds__(256) void k_gemm_out(const _Float16* __restrict__ a16,
                                                  const _Float16* __restrict__ wo16,
                                                  const float* __restrict__ bias,
                                                  float* __restrict__ outp) {
  __shared__ _Float16 ldsA[128 * 40];
  __shared__ _Float16 ldsB[64 * 40];
  __shared__ float    ldsC[8 * 16 * 17];
  const int tid  = threadIdx.x;
  const int wave = tid >> 5, lane = tid & 31;
  const int g = lane >> 4, lm = lane & 15;
  const int m0 = blockIdx.x * 128;
  const int n0 = blockIdx.y * 64;
  v8f acc[4] = {};
  const int arow = tid >> 1, aoff = (tid & 1) * 16;
  const _Float16* asrc = a16 + (size_t)(m0 + arow) * 256 + aoff;
  _Float16* adst = ldsA + arow * 40 + aoff;
  const int brow = tid >> 2, boff = (tid & 3) * 8;
  const _Float16* bsrc = wo16 + (size_t)(n0 + brow) * 256 + boff;
  _Float16* bdst = ldsB + brow * 40 + boff;
  for (int k0 = 0; k0 < 256; k0 += 32) {
    __syncthreads();
    stage16(asrc + k0,     adst);
    stage16(asrc + k0 + 8, adst + 8);
    stage16(bsrc + k0,     bdst);
    stage_wait();
    __syncthreads();
    FragH a;
    const _Float16* ar = ldsA + (wave * 16 + lm) * 40;
    a.q[0] = *(const u32x4*)(ar + g * 8);
    a.q[1] = *(const u32x4*)(ar + 16 + g * 8);
#pragma unroll
    for (int tn = 0; tn < 4; ++tn) {
      FragH bf;
      const _Float16* br = ldsB + (tn * 16 + lm) * 40;
      bf.q[0] = *(const u32x4*)(br + g * 8);
      bf.q[1] = *(const u32x4*)(br + 16 + g * 8);
      acc[tn] = __builtin_amdgcn_wmma_f32_16x16x32_f16(false, a.h, false, bf.h,
                                                       (short)0, acc[tn], false, false);
    }
  }
  const int bb   = m0 >> 12;
  const int posB = m0 & 4095;
  const int rowId = tid >> 1;
  const int wsrc  = rowId >> 4;
  const int d     = rowId & 15;
  const int ho    = (tid & 1) * 8;
#pragma unroll
  for (int tn = 0; tn < 4; ++tn) {
    __syncthreads();
    float* cw = ldsC + wave * (16 * 17);
#pragma unroll
    for (int r = 0; r < 8; ++r)
      cw[lm * 17 + g * 8 + r] = acc[tn][r];
    __syncthreads();
    const int c    = n0 + tn * 16 + d;
    const float bv = bias[c];
    const float* csrc = ldsC + wsrc * (16 * 17) + d * 17 + ho;
    f32x4 o0, o1;
#pragma unroll
    for (int e = 0; e < 4; ++e) { o0[e] = csrc[e] + bv; o1[e] = csrc[4 + e] + bv; }
    float* dst = outp + ((size_t)bb * 256 + c) * 4096 + posB + wsrc * 16 + ho;
    *(f32x4*)dst       = o0;
    *(f32x4*)(dst + 4) = o1;
  }
}

// ---------------------------------------------------------------------------
extern "C" void kernel_launch(void* const* d_in, const int* in_sizes, int n_in,
                              void* d_out, int out_size, void* d_ws, size_t ws_size,
                              hipStream_t stream) {
  (void)in_sizes; (void)n_in; (void)out_size; (void)ws_size;
  const float* x    = (const float*)d_in[0];
  const float* gam  = (const float*)d_in[1];
  const float* bet  = (const float*)d_in[2];
  const float* Wqkv = (const float*)d_in[3];
  const float* temp = (const float*)d_in[4];
  const float* Wout = (const float*)d_in[5];
  const float* bout = (const float*)d_in[6];

  char* ws = (char*)d_ws;
  _Float16* t16   = (_Float16*)ws; ws += (size_t)65536 * 256 * 2;            // 32 MB
  _Float16* w16   = (_Float16*)ws; ws += (size_t)768 * 256 * 2;              // 384 KB
  _Float16* wo16  = (_Float16*)ws; ws += (size_t)256 * 256 * 2;              // 128 KB
  _Float16* qkv16 = (_Float16*)ws; ws += (size_t)3 * 16 * 8 * 32 * 4096 * 2; // 96 MB
  _Float16* out16 = (_Float16*)ws; ws += (size_t)65536 * 256 * 2;            // 32 MB
  float* outp = (float*)d_out;

  k_ln      <<<dim3(16, 128), 256, 0, stream>>>(x, gam, bet, t16);
  k_cvtw    <<<dim3(1024),    256, 0, stream>>>(Wqkv, Wout, w16, wo16);
  k_gemm_qkv<<<dim3(512, 12), 256, 0, stream>>>(t16, w16, qkv16);
  k_l2scale <<<dim3(8192),    256, 0, stream>>>(qkv16, temp);
  k_attn    <<<dim3(128),     128, 0, stream>>>(qkv16, out16);
  k_gemm_out<<<dim3(512, 4),  256, 0, stream>>>(out16, wo16, bout, outp);
}
